// Spa_Extract_Layer_8486855377192
// MI455X (gfx1250) — matrix operand order repaired and checked by reading, not compiled, not run on hardware
//
#include <hip/hip_runtime.h>
#include <hip/hip_bf16.h>
#include <math.h>

// ---------------- problem constants ----------------
#define BB   4
#define CC   10000
#define DMM  128
#define RR   16
#define KK   256
#define HH   8
#define FFF  512
#define NTOK (BB*CC)            // 40000 tokens for the MLP branch
#define FINAL_N (BB*1*CC*DMM)   // 5,120,000
#define TOPK_N  (BB*RR*KK)      // 16,384

// ---------------- workspace layout (float elements) ----------------
#define OFF_CENTER    0                          // 8192
#define OFF_TOPK_IDX  8192                       // 16384 (int view)
#define OFF_TOPK_SIM  24576                      // 16384
#define OFF_COUNT     40960                      // 40000
#define OFF_TOTAL     80960                      // 40000
#define OFF_UPDATE    120960                     // 2,097,152
#define OFF_UPDATED   2218112                    // 5,120,000
#define OFF_LOSS      7338112                    // 2
#define OFF_WPACK     7338240                    // packed f16 weights (aligned)

// packed-weight sub-offsets (in half elements), per attention layer
#define PK_WQ   0
#define PK_WK   16384
#define PK_WV   32768
#define PK_WO   49152
#define PK_W1   65536     // 128x512 = 65536 halves
#define PK_W2   131072    // 512x128 = 65536 halves
#define PK_SIZE 196608    // halves per layer pack

// ---------------- WMMA types & helpers ----------------
typedef __attribute__((ext_vector_type(16))) _Float16 v16h;
typedef __attribute__((ext_vector_type(8)))  float    v8f;
typedef _Float16 half_t;

union AFrag { v16h v; float4 f[2]; };

__device__ __forceinline__ v8f wmma_f16(v16h a, v16h b, v8f c) {
  // D = A(16x32 f16) * B(32x16 f16) + C(16x16 f32)
  return __builtin_amdgcn_wmma_f32_16x16x32_f16(false, a, false, b, (short)0, c,
                                                false, false);
}

// 16-lane-group reductions (rows live within one half of the wave)
__device__ __forceinline__ float rsum16(float x) {
  #pragma unroll
  for (int m = 1; m < 16; m <<= 1) x += __shfl_xor(x, m, 32);
  return x;
}
__device__ __forceinline__ float rmax16(float x) {
  #pragma unroll
  for (int m = 1; m < 16; m <<= 1) x = fmaxf(x, __shfl_xor(x, m, 32));
  return x;
}

// GEMM with packed f16 weights: per 32x16 tile, each lane's 16 halves are
// contiguous -> B fragment = two global b128 loads.
// acc[8] += X(16 rows at m0, Kdim cols, f16 LDS ld=ldx) * W(Kdim x 128 slice, tile col nt0)
__device__ __forceinline__ void gemm_tile_p(const half_t* X, int ldx, int m0,
                                            const half_t* Wp, int Ntiles, int Kdim,
                                            int nt0, v8f acc[8],
                                            int lane, int lo, int hi) {
  for (int kt = 0; kt < (Kdim >> 5); ++kt) {
    AFrag a;
    const half_t* p = X + (size_t)(m0 + lo) * ldx + kt * 32 + hi * 8;
    a.f[0] = *(const float4*)(p);
    a.f[1] = *(const float4*)(p + 16);
    #pragma unroll
    for (int t = 0; t < 8; ++t) {
      const half_t* wp = Wp + (size_t)(kt * Ntiles + nt0 + t) * 512 + lane * 16;
      AFrag b;
      b.f[0] = *(const float4*)(wp);
      b.f[1] = *(const float4*)(wp + 8);
      acc[t] = wmma_f16(a.v, b.v, acc[t]);
    }
  }
}

// residual add + LayerNorm over 128 cols for one 16-row tile held in acc[8].
__device__ __forceinline__ void resid_ln(v8f acc[8], const float* bias,
                                         const half_t* resX, int ldr, int m0,
                                         const float* gamma, const float* beta,
                                         half_t* outL, int ldo,
                                         float* outG, int lo, int hi, int row_limit) {
  float ps[8], ps2[8];
  #pragma unroll
  for (int v = 0; v < 8; ++v) { ps[v] = 0.f; ps2[v] = 0.f; }
  #pragma unroll
  for (int t = 0; t < 8; ++t) {
    int col = t * 16 + lo;
    float bv = bias ? bias[col] : 0.f;
    #pragma unroll
    for (int v = 0; v < 8; ++v) {
      int row = m0 + v + 8 * hi;
      float xv = acc[t][v] + bv + (float)resX[(size_t)row * ldr + col];
      acc[t][v] = xv;
      ps[v] += xv; ps2[v] += xv * xv;
    }
  }
  #pragma unroll
  for (int v = 0; v < 8; ++v) {
    float s  = rsum16(ps[v]);
    float s2 = rsum16(ps2[v]);
    float mean = s * (1.f / 128.f);
    float var  = s2 * (1.f / 128.f) - mean * mean;
    ps[v] = mean;
    ps2[v] = rsqrtf(var + 1e-5f);
  }
  #pragma unroll
  for (int t = 0; t < 8; ++t) {
    int col = t * 16 + lo;
    float g = gamma[col], be = beta[col];
    #pragma unroll
    for (int v = 0; v < 8; ++v) {
      int row = m0 + v + 8 * hi;
      float y = (acc[t][v] - ps[v]) * ps2[v] * g + be;
      if (outL) outL[(size_t)row * ldo + col] = (half_t)y;
      if (outG && row < row_limit) outG[(size_t)row * 128 + col] = y;
    }
  }
}

// ---------------- parameter packs ----------------
struct AttnPk {               // packed f16 weights + fp32 vectors
  const half_t *Wq, *Wk, *Wv, *Wo, *W1, *W2;
  const float  *bq, *bk, *bv, *bo, *b1, *b2, *g1, *be1, *g2, *be2;
};

// ---------------- kernels ----------------

// pack fp32 weight (K x N, row-major) into fragment-major f16 tiles
__global__ void k_packw(const float* W, int K, int N, half_t* dst) {
  int t = blockIdx.x * 256 + threadIdx.x;
  int ntilesN = N >> 4;
  int total = (K >> 5) * ntilesN * 32;
  if (t >= total) return;
  int lane = t & 31; int tile = t >> 5;
  int nt = tile % ntilesN; int kt = tile / ntilesN;
  int lo = lane & 15, hi = lane >> 4;
  half_t* d = dst + (size_t)tile * 512 + lane * 16;
  int n = nt * 16 + lo;
  #pragma unroll
  for (int j = 0; j < 16; ++j) {
    int k = kt * 32 + (j & 7) + ((j >> 3) << 4) + hi * 8;
    d[j] = (half_t)W[(size_t)k * N + n];
  }
}

__global__ void k_zero(float* cnt, float* tot, float* upd, float* loss) {
  size_t i = (size_t)blockIdx.x * blockDim.x + threadIdx.x;
  if (i < (size_t)FINAL_N) upd[i] = 0.f;
  if (i < (size_t)BB * CC) { cnt[i] = 0.f; tot[i] = 0.f; }
  if (i < 2) loss[i] = 0.f;
}

// center = concat(routing_center, tem_routing, routing_spa) @ W_fus + b_fus
__global__ void k_center(const float* rc, const float* tr, const float* rs,
                         const float* Wf, const float* bf, float* center) {
  int i = blockIdx.x * 256 + threadIdx.x;
  if (i >= BB * RR * DMM) return;
  int col = i & 127; int row = i >> 7;
  int b = row / RR, r = row % RR;
  float s = bf[col];
  #pragma unroll 4
  for (int k = 0; k < 128; ++k) {
    float xv = (k < 64) ? rc[r * 64 + k]
             : (k < 96) ? tr[b * 32 + (k - 64)]
                        : rs[r * 32 + (k - 96)];
    s += xv * Wf[k * 128 + col];
  }
  center[i] = s;
}

// per-(b,r): logits -> softmax over C -> top-256 with smallest-index tie-break
__global__ __launch_bounds__(256) void k_simtopk(const float* nr, const float* center,
                          int* tidx, float* tsim, float* cnt, float* tot,
                          float* out_topk) {
  extern __shared__ float sm[];
  float* probs = sm;        // CC
  float* sC = sm + CC;      // 128
  __shared__ float rv[256];
  __shared__ int   ri[256];
  int b = blockIdx.x / RR, r = blockIdx.x % RR;
  int tid = threadIdx.x;
  if (tid < 128) sC[tid] = center[(b * RR + r) * 128 + tid];
  __syncthreads();
  float lmax = -INFINITY;
  for (int c = tid; c < CC; c += 256) {
    const float* x = nr + ((size_t)b * CC + c) * 128;
    float s = 0.f;
    const float4* x4 = (const float4*)x;
    #pragma unroll 8
    for (int q = 0; q < 32; ++q) {
      float4 xv = x4[q];
      s += xv.x * sC[4*q] + xv.y * sC[4*q+1] + xv.z * sC[4*q+2] + xv.w * sC[4*q+3];
    }
    probs[c] = s;
    lmax = fmaxf(lmax, s);
  }
  rv[tid] = lmax; __syncthreads();
  for (int s = 128; s; s >>= 1) { if (tid < s) rv[tid] = fmaxf(rv[tid], rv[tid+s]); __syncthreads(); }
  float M = rv[0]; __syncthreads();
  float lsum = 0.f;
  for (int c = tid; c < CC; c += 256) { float p = __expf(probs[c] - M); probs[c] = p; lsum += p; }
  rv[tid] = lsum; __syncthreads();
  for (int s = 128; s; s >>= 1) { if (tid < s) rv[tid] += rv[tid+s]; __syncthreads(); }
  float invS = 1.f / rv[0]; __syncthreads();

  for (int it = 0; it < KK; ++it) {
    float bv = -1.f; int bi = 0x7fffffff;
    for (int c = tid; c < CC; c += 256) {
      float p = probs[c];
      if (p > bv || (p == bv && c < bi)) { bv = p; bi = c; }
    }
    rv[tid] = bv; ri[tid] = bi; __syncthreads();
    for (int s = 128; s; s >>= 1) {
      if (tid < s) {
        if (rv[tid+s] > rv[tid] || (rv[tid+s] == rv[tid] && ri[tid+s] < ri[tid])) {
          rv[tid] = rv[tid+s]; ri[tid] = ri[tid+s];
        }
      }
      __syncthreads();
    }
    if (tid == 0) {
      int ci = ri[0]; float pv = rv[0] * invS;
      int e = (b * RR + r) * KK + it;
      tidx[e] = ci; tsim[e] = pv; out_topk[e] = (float)ci;
      atomicAdd(&cnt[b * CC + ci], 1.f);
      atomicAdd(&tot[b * CC + ci], pv);
      probs[ci] = -1.f;
    }
    __syncthreads();
  }
}

// full transformer layer ('atten') on gathered tokens: one block per (b,r)
__global__ __launch_bounds__(256) void k_attn1(const float* nr, const int* tidx,
                                               AttnPk P, float* update) {
  extern __shared__ char smraw[];
  half_t* X    = (half_t*)smraw;       // 256*128 (input x, then y1)
  half_t* BUF1 = X + 256 * 128;        // 256*512 (QKV / P scratch / FF hidden)
  half_t* Qb = BUF1;
  half_t* Kb = BUF1 + 32768;
  half_t* Vb = BUF1 + 65536;
  half_t* Ps = BUF1 + 98304;           // 8 waves * 256 halves
  int b = blockIdx.x / RR, r = blockIdx.x % RR;
  int tid = threadIdx.x, lane = tid & 31, wave = tid >> 5;
  int lo = lane & 15, hi = lane >> 4;

  // ---- gather tokens (f32 -> f16) ----
  {
    int j = tid;
    int c = tidx[(b * RR + r) * KK + j];
    const float* src = nr + ((size_t)b * CC + c) * 128;
    half_t* dst = X + j * 128;
    for (int d = 0; d < 128; d += 4) {
      float4 v4 = *(const float4*)(src + d);
      dst[d] = (half_t)v4.x; dst[d+1] = (half_t)v4.y;
      dst[d+2] = (half_t)v4.z; dst[d+3] = (half_t)v4.w;
    }
  }
  __syncthreads();

  // ---- QKV projections (Q pre-scaled by 1/sqrt(dh)) ----
  {
    const half_t* Wp[3] = {P.Wq, P.Wk, P.Wv};
    const float*  bp[3] = {P.bq, P.bk, P.bv};
    half_t* dp[3] = {Qb, Kb, Vb};
    for (int pj = 0; pj < 3; ++pj) {
      float scl = (pj == 0) ? 0.25f : 1.f;
      for (int mt = 0; mt < 2; ++mt) {
        int m0 = wave * 32 + mt * 16;
        v8f acc[8] = {};
        gemm_tile_p(X, 128, m0, Wp[pj], 8, 128, 0, acc, lane, lo, hi);
        #pragma unroll
        for (int t = 0; t < 8; ++t) {
          int col = t * 16 + lo; float bia = bp[pj][col];
          #pragma unroll
          for (int v = 0; v < 8; ++v)
            dp[pj][(m0 + v + 8*hi) * 128 + col] = (half_t)((acc[t][v] + bia) * scl);
        }
      }
    }
  }
  __syncthreads();

  // ---- flash attention: wave == head, dh=16 padded to K=32 ----
  {
    int h = wave;
    half_t* myP = Ps + wave * 256;
    for (int qt = 0; qt < 16; ++qt) {
      int m0 = qt * 16;
      AFrag aq;
      aq.f[0] = *(const float4*)(Qb + (m0 + lo) * 128 + h * 16 + hi * 8);
      aq.f[1] = make_float4(0.f, 0.f, 0.f, 0.f);
      float mi[8], li[8];
      #pragma unroll
      for (int v = 0; v < 8; ++v) { mi[v] = -INFINITY; li[v] = 0.f; }
      v8f O = {};
      for (int kt = 0; kt < 16; ++kt) {
        AFrag bk;
        bk.f[0] = *(const float4*)(Kb + (kt * 16 + lo) * 128 + h * 16 + hi * 8);
        bk.f[1] = make_float4(0.f, 0.f, 0.f, 0.f);
        v8f z = {};
        v8f s = wmma_f16(aq.v, bk.v, z);
        #pragma unroll
        for (int v = 0; v < 8; ++v) {
          float rm = rmax16(s[v]);
          float nm = fmaxf(mi[v], rm);
          float alpha = __expf(mi[v] - nm);
          float p = __expf(s[v] - nm);
          mi[v] = nm;
          li[v] = li[v] * alpha + rsum16(p);
          O[v] *= alpha;
          myP[(v + 8*hi) * 16 + lo] = (half_t)p;
        }
        AFrag ap;
        ap.f[0] = *(const float4*)(myP + lo * 16 + hi * 8);
        ap.f[1] = make_float4(0.f, 0.f, 0.f, 0.f);
        v16h bvv = {};
        #pragma unroll
        for (int j = 0; j < 8; ++j)
          bvv[j] = Vb[(kt * 16 + j + hi * 8) * 128 + h * 16 + lo];
        O = wmma_f16(ap.v, bvv, O);
      }
      #pragma unroll
      for (int v = 0; v < 8; ++v)
        Qb[(m0 + v + 8*hi) * 128 + h * 16 + lo] = (half_t)(O[v] / li[v]);
    }
  }
  __syncthreads();

  // ---- Wo + residual + LN1 (y1 -> X) ----
  for (int mt = 0; mt < 2; ++mt) {
    int m0 = wave * 32 + mt * 16;
    v8f acc[8] = {};
    gemm_tile_p(Qb, 128, m0, P.Wo, 8, 128, 0, acc, lane, lo, hi);
    resid_ln(acc, P.bo, X, 128, m0, P.g1, P.be1, X, 128, nullptr, lo, hi, 256);
  }
  __syncthreads();

  // ---- FF1: relu(y1 @ W1 + b1) -> BUF1 (256x512) ----
  for (int mt = 0; mt < 2; ++mt) {
    int m0 = wave * 32 + mt * 16;
    for (int nc = 0; nc < 4; ++nc) {
      v8f acc[8] = {};
      gemm_tile_p(X, 128, m0, P.W1, 32, 128, nc * 8, acc, lane, lo, hi);
      #pragma unroll
      for (int t = 0; t < 8; ++t) {
        int col = nc * 128 + t * 16 + lo; float bia = P.b1[col];
        #pragma unroll
        for (int v = 0; v < 8; ++v)
          BUF1[(size_t)(m0 + v + 8*hi) * 512 + col] = (half_t)fmaxf(acc[t][v] + bia, 0.f);
      }
    }
  }
  // ---- FF2 + residual + LN2 -> update (f32, global) ----
  for (int mt = 0; mt < 2; ++mt) {
    int m0 = wave * 32 + mt * 16;
    v8f acc[8] = {};
    gemm_tile_p(BUF1, 512, m0, P.W2, 8, 512, 0, acc, lane, lo, hi);
    float* outG = update + ((size_t)(b * RR + r) * KK) * 128;
    resid_ln(acc, P.b2, X, 128, m0, P.g2, P.be2, nullptr, 0, outG, lo, hi, 256);
  }
}

// 'mlp' layer on all B*C tokens -> writes 'unselect' straight into d_out
__global__ __launch_bounds__(256) void k_mlp(const float* xg, AttnPk P, float* outg) {
  extern __shared__ char smraw[];
  half_t* X  = (half_t*)smraw;     // 128*128 (x, then y1)
  half_t* Hh = X + 128 * 128;      // 128*512 (t, then hidden)
  int tid = threadIdx.x, lane = tid & 31, wave = tid >> 5;
  int lo = lane & 15, hi = lane >> 4;
  int tok0 = blockIdx.x * 128;
  int limit = NTOK - tok0;

  for (int i = tid; i < 128 * 128; i += 256) {
    int row = i >> 7, col = i & 127;
    float v = (row < limit) ? xg[((size_t)tok0 + row) * 128 + col] : 0.f;
    X[i] = (half_t)v;
  }
  __syncthreads();

  int m0 = wave * 16;
  { // t = x @ Wv + bv -> Hh (cols 0..127, ld 512)
    v8f acc[8] = {};
    gemm_tile_p(X, 128, m0, P.Wv, 8, 128, 0, acc, lane, lo, hi);
    #pragma unroll
    for (int t = 0; t < 8; ++t) {
      int col = t * 16 + lo; float bia = P.bv[col];
      #pragma unroll
      for (int v = 0; v < 8; ++v)
        Hh[(size_t)(m0 + v + 8*hi) * 512 + col] = (half_t)(acc[t][v] + bia);
    }
  }
  { // u = t @ Wo + bo; y1 = LN(x + u) -> X
    v8f acc[8] = {};
    gemm_tile_p(Hh, 512, m0, P.Wo, 8, 128, 0, acc, lane, lo, hi);
    resid_ln(acc, P.bo, X, 128, m0, P.g1, P.be1, X, 128, nullptr, lo, hi, 128);
  }
  for (int nc = 0; nc < 4; ++nc) { // h = relu(y1 @ W1 + b1) -> Hh
    v8f acc[8] = {};
    gemm_tile_p(X, 128, m0, P.W1, 32, 128, nc * 8, acc, lane, lo, hi);
    #pragma unroll
    for (int t = 0; t < 8; ++t) {
      int col = nc * 128 + t * 16 + lo; float bia = P.b1[col];
      #pragma unroll
      for (int v = 0; v < 8; ++v)
        Hh[(size_t)(m0 + v + 8*hi) * 512 + col] = (half_t)fmaxf(acc[t][v] + bia, 0.f);
    }
  }
  { // v = h @ W2 + b2; out = LN(y1 + v) -> global
    v8f acc[8] = {};
    gemm_tile_p(Hh, 512, m0, P.W2, 8, 512, 0, acc, lane, lo, hi);
    resid_ln(acc, P.b2, X, 128, m0, P.g2, P.be2, nullptr, 0,
             outg + (size_t)tok0 * 128, lo, hi, limit);
  }
}

// scatter: updated[b,c,:] += (sim/total) * update[e,:]
__global__ void k_scatter(const int* tidx, const float* tsim, const float* tot,
                          const float* cnt, const float* update, float* updated) {
  int e = blockIdx.x * 256 + threadIdx.x;
  if (e >= TOPK_N) return;
  int b = e / (RR * KK);
  int c = tidx[e];
  float T = tot[b * CC + c];
  float w = (T > 0.f) ? (tsim[e] / T) : (1.f / fmaxf(cnt[b * CC + c], 1.f));
  const float* u = update + (size_t)e * 128;
  float* dst = updated + ((size_t)b * CC + c) * 128;
  for (int d = 0; d < 128; ++d) atomicAdd(dst + d, w * u[d]);
}

// final = selected ? updated : unselect (unselect already in d_out)
__global__ void k_select(const float* cnt, const float* updated, float* out) {
  size_t i = (size_t)blockIdx.x * 256 + threadIdx.x;
  if (i >= (size_t)FINAL_N) return;
  size_t row = i >> 7;
  if (cnt[row] > 0.f) out[i] = updated[i];
}

__global__ void k_losscenter(const float* center, float* loss) {
  __shared__ float zc[64 * 128];
  __shared__ float zn[64];
  int tid = threadIdx.x;
  for (int i = tid; i < 64 * 128; i += 256) zc[i] = center[i];
  __syncthreads();
  if (tid < 64) {
    float s = 0.f;
    for (int d = 0; d < 128; ++d) { float v = zc[tid * 128 + d]; s += v * v; }
    zn[tid] = fmaxf(sqrtf(s), 1e-12f);
  }
  __syncthreads();
  if (tid < 64) {
    int b = tid >> 4, i = tid & 15;
    float neg = 0.f;
    for (int j = 0; j < 16; ++j) {
      if (j == i) continue;
      float dot = 0.f;
      for (int d = 0; d < 128; ++d)
        dot += zc[(b * 16 + i) * 128 + d] * zc[(b * 16 + j) * 128 + d];
      neg += __expf(dot / (zn[b * 16 + i] * zn[b * 16 + j]) * (1.f / 0.3f));
    }
    float de = __expf(1.f / 0.3f);
    float term = -logf(de / (de + neg) + 1e-8f);
    atomicAdd(&loss[0], term * (1.f / 64.f));
  }
}

__global__ void k_losscluster(const float* center, const float* update, float* loss) {
  extern __shared__ float sm[];
  float* lg = sm;           // RR*KK = 4096 logits
  float* sc = sm + RR * KK; // 128
  __shared__ float red[256];
  __shared__ float s_cn;
  int b = blockIdx.x / RR, r = blockIdx.x % RR;
  int tid = threadIdx.x;
  if (tid < 128) sc[tid] = center[(b * RR + r) * 128 + tid];
  __syncthreads();
  if (tid == 0) {
    float s = 0.f;
    for (int d = 0; d < 128; ++d) s += sc[d] * sc[d];
    s_cn = fmaxf(sqrtf(s), 1e-12f);
  }
  __syncthreads();
  float cn = s_cn;
  for (int n = tid; n < RR * KK; n += 256) {
    const float* x = update + ((size_t)b * RR * KK + n) * 128;
    float dot = 0.f, xs = 0.f;
    for (int d = 0; d < 128; ++d) { float xv = x[d]; dot += xv * sc[d]; xs += xv * xv; }
    lg[n] = dot / (cn * fmaxf(sqrtf(xs), 1e-12f)) * 10.f;  // /temp 0.1
  }
  __syncthreads();
  float lm = -INFINITY;
  for (int n = tid; n < RR * KK; n += 256) lm = fmaxf(lm, lg[n]);
  red[tid] = lm; __syncthreads();
  for (int s = 128; s; s >>= 1) { if (tid < s) red[tid] = fmaxf(red[tid], red[tid+s]); __syncthreads(); }
  float Mall = red[0]; __syncthreads();
  float ls = 0.f;
  for (int n = tid; n < RR * KK; n += 256) ls += __expf(lg[n] - Mall);
  red[tid] = ls; __syncthreads();
  for (int s = 128; s; s >>= 1) { if (tid < s) red[tid] += red[tid+s]; __syncthreads(); }
  float lp_all = Mall + logf(red[0]); __syncthreads();
  float pv = lg[r * KK + tid];
  red[tid] = pv; __syncthreads();
  for (int s = 128; s; s >>= 1) { if (tid < s) red[tid] = fmaxf(red[tid], red[tid+s]); __syncthreads(); }
  float Mp = red[0]; __syncthreads();
  red[tid] = __expf(pv - Mp); __syncthreads();
  for (int s = 128; s; s >>= 1) { if (tid < s) red[tid] += red[tid+s]; __syncthreads(); }
  if (tid == 0) {
    float lp_pos = Mp + logf(red[0]);
    atomicAdd(&loss[1], -(lp_pos - lp_all) * (1.f / 64.f));
  }
}

__global__ void k_losswrite(const float* loss, float* out) { out[0] = loss[0] + loss[1]; }

// ---------------- launch ----------------
extern "C" void kernel_launch(void* const* d_in, const int* in_sizes, int n_in,
                              void* d_out, int out_size, void* d_ws, size_t ws_size,
                              hipStream_t stream) {
  (void)in_sizes; (void)n_in; (void)out_size; (void)ws_size;
  const float* node = (const float*)d_in[0];
  const float* tem  = (const float*)d_in[1];
  const float* rc   = (const float*)d_in[2];
  const float* rs   = (const float*)d_in[3];
  const float* Wf   = (const float*)d_in[4];
  const float* bf   = (const float*)d_in[5];
  auto F = [&](int i) { return (const float*)d_in[i]; };

  float* ws = (float*)d_ws;
  float* wsCenter  = ws + OFF_CENTER;
  int*   wsIdx     = (int*)(ws + OFF_TOPK_IDX);
  float* wsSim     = ws + OFF_TOPK_SIM;
  float* wsCnt     = ws + OFF_COUNT;
  float* wsTot     = ws + OFF_TOTAL;
  float* wsUpd     = ws + OFF_UPDATE;
  float* wsUpdated = ws + OFF_UPDATED;
  float* wsLoss    = ws + OFF_LOSS;
  half_t* pk1 = (half_t*)(ws + OFF_WPACK);
  half_t* pk2 = pk1 + PK_SIZE;
  float* out = (float*)d_out;

  // packed-weight parameter packs (attn1 at d_in[6..21], attn2 at [22..37];
  // dict order: Wq,bq,Wk,bk,Wv,bv,Wo,bo,W1,b1,W2,b2,g1,be1,g2,be2)
  AttnPk A1 { pk1 + PK_WQ, pk1 + PK_WK, pk1 + PK_WV, pk1 + PK_WO,
              pk1 + PK_W1, pk1 + PK_W2,
              F(7), F(9), F(11), F(13), F(15), F(17), F(18), F(19), F(20), F(21) };
  AttnPk A2 { pk2 + PK_WQ, pk2 + PK_WK, pk2 + PK_WV, pk2 + PK_WO,
              pk2 + PK_W1, pk2 + PK_W2,
              F(23), F(25), F(27), F(29), F(31), F(33), F(34), F(35), F(36), F(37) };

  // one-shot weight packing (f32 -> fragment-major f16)
  k_packw<<<4,  256, 0, stream>>>(F(6),  128, 128, pk1 + PK_WQ);
  k_packw<<<4,  256, 0, stream>>>(F(8),  128, 128, pk1 + PK_WK);
  k_packw<<<4,  256, 0, stream>>>(F(10), 128, 128, pk1 + PK_WV);
  k_packw<<<4,  256, 0, stream>>>(F(12), 128, 128, pk1 + PK_WO);
  k_packw<<<16, 256, 0, stream>>>(F(14), 128, 512, pk1 + PK_W1);
  k_packw<<<16, 256, 0, stream>>>(F(16), 512, 128, pk1 + PK_W2);
  k_packw<<<4,  256, 0, stream>>>(F(22), 128, 128, pk2 + PK_WQ);
  k_packw<<<4,  256, 0, stream>>>(F(24), 128, 128, pk2 + PK_WK);
  k_packw<<<4,  256, 0, stream>>>(F(26), 128, 128, pk2 + PK_WV);
  k_packw<<<4,  256, 0, stream>>>(F(28), 128, 128, pk2 + PK_WO);
  k_packw<<<16, 256, 0, stream>>>(F(30), 128, 512, pk2 + PK_W1);
  k_packw<<<16, 256, 0, stream>>>(F(32), 512, 128, pk2 + PK_W2);

  k_zero<<<(FINAL_N + 255) / 256, 256, 0, stream>>>(wsCnt, wsTot, wsUpdated, wsLoss);
  k_center<<<(BB * RR * DMM + 255) / 256, 256, 0, stream>>>(rc, tem, rs, Wf, bf, wsCenter);
  k_simtopk<<<BB * RR, 256, (CC + 128) * sizeof(float), stream>>>(
      node, wsCenter, wsIdx, wsSim, wsCnt, wsTot, out + FINAL_N);
  k_attn1<<<BB * RR, 256, (256 * 128 + 256 * 512) * sizeof(half_t), stream>>>(
      node, wsIdx, A1, wsUpd);
  k_mlp<<<(NTOK + 127) / 128, 256, (128 * 128 + 128 * 512) * sizeof(half_t), stream>>>(
      node, A2, out);
  k_scatter<<<(TOPK_N + 255) / 256, 256, 0, stream>>>(wsIdx, wsSim, wsTot, wsCnt,
                                                      wsUpd, wsUpdated);
  k_select<<<(FINAL_N + 255) / 256, 256, 0, stream>>>(wsCnt, wsUpdated, out);
  k_losscenter<<<1, 256, 0, stream>>>(wsCenter, wsLoss);
  k_losscluster<<<BB * RR, 256, (RR * KK + 128) * sizeof(float), stream>>>(
      wsCenter, wsUpd, wsLoss);
  k_losswrite<<<1, 1, 0, stream>>>(wsLoss, out + FINAL_N + TOPK_N);
}
